// structure_encoder_19619410608623
// MI455X (gfx1250) — compile-verified, matrix operand order checked by reading, not run on hardware
//
#include <hip/hip_runtime.h>
#include <hip/hip_bf16.h>
#include <math.h>

#define DEVI __device__ __forceinline__

typedef __attribute__((ext_vector_type(16))) __bf16 v16bf;
typedef __attribute__((ext_vector_type(8)))  float  v8f;

static constexpr int Nn = 20000;
static constexpr int Ne = 600000;

DEVI unsigned short f2bf(float f) {
  unsigned int u = __float_as_uint(f);
  u += 0x7FFFu + ((u >> 16) & 1u);            // round-to-nearest-even
  return (unsigned short)(u >> 16);
}
DEVI float sigm(float x) { return 1.f / (1.f + __expf(-x)); }
DEVI float clipn(float s) { return fmaxf(s, 1e-8f); }
DEVI float atomic_add_f(float* p, float v) {
  return __hip_atomic_fetch_add(p, v, __ATOMIC_RELAXED, __HIP_MEMORY_SCOPE_AGENT);
}
DEVI v8f zero8() {
  v8f z;
#pragma unroll
  for (int i = 0; i < 8; ++i) z[i] = 0.f;
  return z;
}

// Generic (flat) pointer to a __shared__ object: low 32 bits are the LDS offset.
DEVI unsigned lds_addr_of(const void* p) { return (unsigned)(unsigned long long)p; }

// CDNA5 async gather: per-lane global 16B -> per-lane LDS 16B, tracked on ASYNCcnt.
DEVI void async_g2l_b128(unsigned ldsaddr, const void* gaddr) {
  asm volatile("global_load_async_to_lds_b128 %0, %1, off"
               :: "v"(ldsaddr), "v"((unsigned long long)(size_t)gaddr)
               : "memory");
}
DEVI void wait_async0() {
#if __has_builtin(__builtin_amdgcn_s_wait_asynccnt)
  __builtin_amdgcn_s_wait_asynccnt(0);
#else
  asm volatile("s_wait_asynccnt 0x0" ::: "memory");
#endif
}

// A fragment: 16x32 bf16, row m = lane%16, K pattern per CDNA5 ISA layout.
// row points at the start of row m (bf16/ushort elements, K contiguous).
DEVI v16bf load_a_frag(const unsigned short* row, int kt, int lane) {
  int base = kt * 32 + ((lane & 16) ? 8 : 0);
  union { unsigned int u[8]; v16bf v; } f;
#pragma unroll
  for (int j = 0; j < 8; ++j) {
    int k = base + ((j < 4) ? (2 * j) : (16 + 2 * (j - 4)));
    f.u[j] = *(const unsigned int*)(row + k);
  }
  return f.v;
}

// B fragment: pre-packed so each lane's 16 bf16 are contiguous (32B).
DEVI v16bf load_b_frag(const unsigned short* Bp, int kt, int nt, int NT, int lane) {
  const unsigned short* p = Bp + ((size_t)(kt * NT + nt) * 512) + lane * 16;
  union { uint4 q[2]; v16bf v; } f;
  f.q[0] = ((const uint4*)p)[0];
  f.q[1] = ((const uint4*)p)[1];
  return f.v;
}

// Preload all 7 B fragments of a K-step, then a 7-WMMA burst (one wait per K-step).
template <int KT>
DEVI void gemm7(const unsigned short* arow, const unsigned short* Bp, int lane, v8f acc[7]) {
#pragma unroll
  for (int nt = 0; nt < 7; ++nt) acc[nt] = zero8();
  for (int kt = 0; kt < KT; ++kt) {
    v16bf a = load_a_frag(arow, kt, lane);
    v16bf bb[7];
#pragma unroll
    for (int nt = 0; nt < 7; ++nt) bb[nt] = load_b_frag(Bp, kt, nt, 7, lane);
#pragma unroll
    for (int nt = 0; nt < 7; ++nt) {
      acc[nt] = __builtin_amdgcn_wmma_f32_16x16x32_bf16(false, a, false, bb[nt], (short)0,
                                                        acc[nt], false, false);
    }
  }
}

// ---------- weight repack: f32 [Nout x K] row-major -> bf16 WMMA-B tiles ----------
__global__ void pack_b_kernel(const float* __restrict__ W, int Nout, int K, int KT, int NT,
                              unsigned short* __restrict__ dst) {
  int total = KT * NT * 512;
  for (int idx = blockIdx.x * blockDim.x + threadIdx.x; idx < total;
       idx += gridDim.x * blockDim.x) {
    int i = idx & 15;
    int lane = (idx >> 4) & 31;
    int tile = idx >> 9;
    int nt = tile % NT, kt = tile / NT;
    int n = nt * 16 + (lane & 15);
    int k = kt * 32 + ((lane & 16) ? 8 : 0) + (i < 8 ? i : i + 8);
    float v = (n < Nout && k < K) ? W[(size_t)n * K + k] : 0.f;
    dst[idx] = f2bf(v);
  }
}

// ---------- degree ----------
__global__ void deg_count_kernel(const int* __restrict__ eidx, unsigned int* __restrict__ cnt,
                                 int E) {
  int e = blockIdx.x * blockDim.x + threadIdx.x;
  if (e < E) atomicAdd(&cnt[eidx[E + e]], 1u);
}
__global__ void deg_inv_kernel(const unsigned int* __restrict__ cnt, float* __restrict__ inv,
                               int N) {
  int n = blockIdx.x * blockDim.x + threadIdx.x;
  if (n < N) inv[n] = 1.f / fmaxf((float)cnt[n], 1.f);
}

// ---------- node embedding: LN + GVP(6,3 -> 100,16) ----------
__global__ void embed_nodes_kernel(const float* __restrict__ ns, const float* __restrict__ nv,
                                   const float* __restrict__ lng, const float* __restrict__ lnb,
                                   const float* __restrict__ wh, const float* __restrict__ wsb,
                                   const float* __restrict__ wsw, const float* __restrict__ wv,
                                   float* __restrict__ hs, float* __restrict__ hv, int N) {
  int n = blockIdx.x * blockDim.x + threadIdx.x;
  if (n >= N) return;
  float s[6], mu = 0.f;
  for (int i = 0; i < 6; ++i) { s[i] = ns[(size_t)n * 6 + i]; mu += s[i]; }
  mu /= 6.f;
  float var = 0.f;
  for (int i = 0; i < 6; ++i) { float d = s[i] - mu; var += d * d; }
  float istd = rsqrtf(var / 6.f + 1e-5f);
  for (int i = 0; i < 6; ++i) s[i] = (s[i] - mu) * istd * lng[i] + lnb[i];
  float v[3][3], msq = 0.f;
  for (int j = 0; j < 3; ++j) {
    float ss = 0.f;
    for (int c = 0; c < 3; ++c) { v[j][c] = nv[(size_t)n * 9 + j * 3 + c]; ss += v[j][c] * v[j][c]; }
    msq += clipn(ss);
  }
  float vinv = rsqrtf(msq / 3.f);
  for (int j = 0; j < 3; ++j)
    for (int c = 0; c < 3; ++c) v[j][c] *= vinv;
  float vh[16][3], sin[22];
  for (int i = 0; i < 6; ++i) sin[i] = s[i];
  for (int h = 0; h < 16; ++h) {
    float x = 0, y = 0, z = 0;
    for (int j = 0; j < 3; ++j) { float w = wh[h * 3 + j]; x += w * v[j][0]; y += w * v[j][1]; z += w * v[j][2]; }
    vh[h][0] = x; vh[h][1] = y; vh[h][2] = z;
    sin[6 + h] = sqrtf(clipn(x * x + y * y + z * z));
  }
  for (int o = 0; o < 100; ++o) {
    float a = wsb[o];
    const float* w = wsw + o * 22;
    for (int k = 0; k < 22; ++k) a += w[k] * sin[k];
    hs[(size_t)n * 100 + o] = a;
  }
  for (int o = 0; o < 16; ++o) {
    float x = 0, y = 0, z = 0;
    const float* w = wv + o * 16;
    for (int h = 0; h < 16; ++h) { float ww = w[h]; x += ww * vh[h][0]; y += ww * vh[h][1]; z += ww * vh[h][2]; }
    hv[(size_t)n * 48 + o * 3] = x; hv[(size_t)n * 48 + o * 3 + 1] = y; hv[(size_t)n * 48 + o * 3 + 2] = z;
  }
}

// ---------- edge embedding: LN + GVP(32,1 -> 32,1) ----------
__global__ void embed_edges_kernel(const float* __restrict__ es, const float* __restrict__ ev,
                                   const float* __restrict__ lnb, const float* __restrict__ lng,
                                   const float* __restrict__ wh, const float* __restrict__ wsb,
                                   const float* __restrict__ wsw, const float* __restrict__ wv,
                                   float* __restrict__ es2, float* __restrict__ ev2, int E) {
  int e = blockIdx.x * blockDim.x + threadIdx.x;
  if (e >= E) return;
  float s[32], mu = 0.f;
  for (int i = 0; i < 32; ++i) { s[i] = es[(size_t)e * 32 + i]; mu += s[i]; }
  mu /= 32.f;
  float var = 0.f;
  for (int i = 0; i < 32; ++i) { float d = s[i] - mu; var += d * d; }
  float istd = rsqrtf(var / 32.f + 1e-5f);
  for (int i = 0; i < 32; ++i) s[i] = (s[i] - mu) * istd * lng[i] + lnb[i];
  float v[3];
  float ss = 0.f;
  for (int c = 0; c < 3; ++c) { v[c] = ev[(size_t)e * 3 + c]; ss += v[c] * v[c]; }
  float vinv = rsqrtf(clipn(ss));
  for (int c = 0; c < 3; ++c) v[c] *= vinv;
  float wh00 = wh[0];
  float vhv[3] = {wh00 * v[0], wh00 * v[1], wh00 * v[2]};
  float vnh = sqrtf(clipn(vhv[0] * vhv[0] + vhv[1] * vhv[1] + vhv[2] * vhv[2]));
  float sin[33];
  for (int i = 0; i < 32; ++i) sin[i] = s[i];
  sin[32] = vnh;
  for (int o = 0; o < 32; ++o) {
    float a = wsb[o];
    const float* w = wsw + o * 33;
    for (int k = 0; k < 33; ++k) a += w[k] * sin[k];
    es2[(size_t)e * 32 + o] = a;
  }
  float wv00 = wv[0];
  for (int c = 0; c < 3; ++c) ev2[(size_t)e * 3 + c] = wv00 * vhv[c];
}

// ---------- per-layer edge message: GVPx3, async LDS gather + WMMA scalar GEMMs ----------
__global__ __launch_bounds__(64) void edge_msg_kernel(
    const float* __restrict__ hs, const float* __restrict__ hv, const float* __restrict__ es2,
    const float* __restrict__ ev2, const int* __restrict__ eidx, const float* __restrict__ wh0,
    const float* __restrict__ wv0, const float* __restrict__ wh1, const float* __restrict__ wv1,
    const float* __restrict__ wh2, const float* __restrict__ wv2, const float* __restrict__ b0,
    const float* __restrict__ b1, const float* __restrict__ b2,
    const unsigned short* __restrict__ Bp0, const unsigned short* __restrict__ Bp1,
    const unsigned short* __restrict__ Bp2, float* __restrict__ acc_s, float* __restrict__ acc_v,
    int E) {
  __shared__ unsigned short sA0[32 * 288];                    // sin0 bf16 (K 265->288)
  __shared__ unsigned short sA1[32 * 128];                    // sin1 bf16 (116->128)
  __shared__ unsigned short sA2[32 * 128];                    // sin2 bf16
  __shared__ __attribute__((aligned(16))) float sMV[32 * 104];// hv[src] @0..47, hv[dst] @52..99 (16B-aligned blocks)
  __shared__ float sVH[32 * 99];                              // vh0 / vh1 / vh2
  __shared__ __attribute__((aligned(16))) float sEV[96];      // ev for the 32 edges
  __shared__ int sSrc[32];
  __shared__ int sDst[32];

  const int tid = threadIdx.x;
  const int ebase = blockIdx.x * 32;
  const int el = tid >> 1;
  const int t = tid & 1;
  const int e = ebase + el;

  if (t == 0) { sSrc[el] = eidx[e]; sDst[el] = eidx[E + e]; }
  __syncthreads();
  const int src = sSrc[el], dst = sDst[el];

  // ---- async gather (ASYNCcnt): per-lane global 16B chunks -> LDS ----
  // chunks 0..383: hv[src] (32 edges x 12 chunks), 384..767: hv[dst], 768..791: ev stream
  for (int c = tid; c < 792; c += 64) {
    unsigned lds;
    const char* g;
    if (c < 768) {
      int cc = (c < 384) ? c : c - 384;
      int el2 = cc / 12, j = cc - el2 * 12;
      int node = (c < 384) ? sSrc[el2] : sDst[el2];
      g = (const char*)hv + (size_t)node * 192 + j * 16;
      lds = lds_addr_of(&sMV[el2 * 104]) + ((c < 384) ? 0 : 208) + j * 16;
    } else {
      int cc = c - 768;
      g = (const char*)ev2 + (size_t)ebase * 12 + cc * 16;
      lds = lds_addr_of(&sEV[0]) + cc * 16;
    }
    async_g2l_b128(lds, g);
  }

  __builtin_prefetch(hs + (size_t)src * 100, 0, 0);
  __builtin_prefetch(hs + (size_t)dst * 100, 0, 0);

  float* mv = &sMV[el * 104];
  float* vh = &sVH[el * 99];
  unsigned short* a0 = &sA0[el * 288];
  unsigned short* a1 = &sA1[el * 128];
  unsigned short* a2 = &sA2[el * 128];

  // overlap with async: gather ms = [hs[src](100), es(32), hs[dst](100)] -> bf16
  for (int f = t; f < 232; f += 2) {
    float val;
    if (f < 100)      val = hs[(size_t)src * 100 + f];
    else if (f < 132) val = es2[(size_t)e * 32 + (f - 100)];
    else              val = hs[(size_t)dst * 100 + (f - 132)];
    a0[f] = f2bf(val);
  }
  wait_async0();
  __syncthreads();

  // vh0 = wh0[33x33] @ mv ; vn0 -> a0 cols 232..264 ; pad 265..287
  for (int h = t; h < 33; h += 2) {
    float x = 0, y = 0, z = 0;
    const float* w = wh0 + h * 33;
    for (int v = 0; v < 16; ++v) { float ww = w[v]; x += ww * mv[v * 3]; y += ww * mv[v * 3 + 1]; z += ww * mv[v * 3 + 2]; }
    { float ww = w[16]; x += ww * sEV[el * 3]; y += ww * sEV[el * 3 + 1]; z += ww * sEV[el * 3 + 2]; }
    for (int v = 17; v < 33; ++v) {
      int b = 52 + 3 * (v - 17);
      float ww = w[v];
      x += ww * mv[b]; y += ww * mv[b + 1]; z += ww * mv[b + 2];
    }
    vh[h * 3] = x; vh[h * 3 + 1] = y; vh[h * 3 + 2] = z;
    a0[232 + h] = f2bf(sqrtf(clipn(x * x + y * y + z * z)));
  }
  for (int f = 265 + t; f < 288; f += 2) a0[f] = 0;
  __syncthreads();

  // v0 = sigmoid-gated wv0[16x33] @ vh0 -> reuse sMV front
  for (int o = t; o < 16; o += 2) {
    float x = 0, y = 0, z = 0;
    const float* w = wv0 + o * 33;
    for (int h = 0; h < 33; ++h) { float ww = w[h]; x += ww * vh[h * 3]; y += ww * vh[h * 3 + 1]; z += ww * vh[h * 3 + 2]; }
    float g = sigm(sqrtf(clipn(x * x + y * y + z * z)));
    mv[o * 3] = x * g; mv[o * 3 + 1] = y * g; mv[o * 3 + 2] = z * g;
  }
  __syncthreads();

  // vh1 = wh1[16x16] @ v0 ; vn1 -> a1 cols 100..115 ; pad 116..127
  for (int h = t; h < 16; h += 2) {
    float x = 0, y = 0, z = 0;
    const float* w = wh1 + h * 16;
    for (int v = 0; v < 16; ++v) { float ww = w[v]; x += ww * mv[v * 3]; y += ww * mv[v * 3 + 1]; z += ww * mv[v * 3 + 2]; }
    vh[h * 3] = x; vh[h * 3 + 1] = y; vh[h * 3 + 2] = z;
    a1[100 + h] = f2bf(sqrtf(clipn(x * x + y * y + z * z)));
  }
  for (int f = 116 + t; f < 128; f += 2) a1[f] = 0;
  __syncthreads();

  // v1 = gated wv1 @ vh1 -> sMV
  for (int o = t; o < 16; o += 2) {
    float x = 0, y = 0, z = 0;
    const float* w = wv1 + o * 16;
    for (int h = 0; h < 16; ++h) { float ww = w[h]; x += ww * vh[h * 3]; y += ww * vh[h * 3 + 1]; z += ww * vh[h * 3 + 2]; }
    float g = sigm(sqrtf(clipn(x * x + y * y + z * z)));
    mv[o * 3] = x * g; mv[o * 3 + 1] = y * g; mv[o * 3 + 2] = z * g;
  }
  __syncthreads();

  // vh2 = wh2 @ v1 ; vn2 -> a2 cols 100..115 ; pad
  for (int h = t; h < 16; h += 2) {
    float x = 0, y = 0, z = 0;
    const float* w = wh2 + h * 16;
    for (int v = 0; v < 16; ++v) { float ww = w[v]; x += ww * mv[v * 3]; y += ww * mv[v * 3 + 1]; z += ww * mv[v * 3 + 2]; }
    vh[h * 3] = x; vh[h * 3 + 1] = y; vh[h * 3 + 2] = z;
    a2[100 + h] = f2bf(sqrtf(clipn(x * x + y * y + z * z)));
  }
  for (int f = 116 + t; f < 128; f += 2) a2[f] = 0;
  __syncthreads();

  // v2 = wv2 @ vh2 (no gate) -> atomic scatter into acc_v
  for (int o = t; o < 16; o += 2) {
    float x = 0, y = 0, z = 0;
    const float* w = wv2 + o * 16;
    for (int h = 0; h < 16; ++h) { float ww = w[h]; x += ww * vh[h * 3]; y += ww * vh[h * 3 + 1]; z += ww * vh[h * 3 + 2]; }
    atomic_add_f(&acc_v[(size_t)dst * 48 + o * 3], x);
    atomic_add_f(&acc_v[(size_t)dst * 48 + o * 3 + 1], y);
    atomic_add_f(&acc_v[(size_t)dst * 48 + o * 3 + 2], z);
  }
  __syncthreads();

  // ================= WMMA phase (2 waves, 16 edges per wave) =================
  const int lane = tid & 31;
  const int wv_id = tid >> 5;
  const int mrow = wv_id * 16 + (lane & 15);
  v8f acc[7];

  // GEMM0: s0 = relu(ws0 @ sin0 + b0), K=288 (9 steps), N=112
  gemm7<9>(&sA0[mrow * 288], Bp0, lane, acc);
#pragma unroll
  for (int nt = 0; nt < 7; ++nt) {
    int n = nt * 16 + (lane & 15);
    float bs = (n < 100) ? b0[n] : 0.f;
#pragma unroll
    for (int r = 0; r < 8; ++r) {
      int m = wv_id * 16 + r + ((lane & 16) ? 8 : 0);
      if (n < 100) sA1[m * 128 + n] = f2bf(fmaxf(acc[nt][r] + bs, 0.f));
    }
  }
  __syncthreads();

  // GEMM1: s1 = relu(ws1 @ sin1 + b1), K=128 (4 steps)
  gemm7<4>(&sA1[mrow * 128], Bp1, lane, acc);
#pragma unroll
  for (int nt = 0; nt < 7; ++nt) {
    int n = nt * 16 + (lane & 15);
    float bs = (n < 100) ? b1[n] : 0.f;
#pragma unroll
    for (int r = 0; r < 8; ++r) {
      int m = wv_id * 16 + r + ((lane & 16) ? 8 : 0);
      if (n < 100) sA2[m * 128 + n] = f2bf(fmaxf(acc[nt][r] + bs, 0.f));
    }
  }
  __syncthreads();

  // GEMM2: s2 = ws2 @ sin2 + b2 (no act) -> atomic scatter (segment sum)
  gemm7<4>(&sA2[mrow * 128], Bp2, lane, acc);
#pragma unroll
  for (int nt = 0; nt < 7; ++nt) {
    int n = nt * 16 + (lane & 15);
    if (n < 100) {
      float bs = b2[n];
#pragma unroll
      for (int r = 0; r < 8; ++r) {
        int m = wv_id * 16 + r + ((lane & 16) ? 8 : 0);
        atomic_add_f(&acc_s[(size_t)sDst[m] * 100 + n], acc[nt][r] + bs);
      }
    }
  }
}

// ---------- node residual + LN0 + ff0 prep ----------
__global__ void node_ln0_kernel(const float* __restrict__ acc_s, const float* __restrict__ acc_v,
                                const float* __restrict__ invdeg, const float* __restrict__ lng,
                                const float* __restrict__ lnb, const float* __restrict__ whf0,
                                float* __restrict__ hs, float* __restrict__ hv,
                                float* __restrict__ vhf, unsigned short* __restrict__ aff0,
                                int N) {
  int n = blockIdx.x * blockDim.x + threadIdx.x;
  if (n >= N) return;
  float inv = invdeg[n];
  float s[100], mu = 0.f;
  for (int o = 0; o < 100; ++o) { s[o] = hs[(size_t)n * 100 + o] + acc_s[(size_t)n * 100 + o] * inv; mu += s[o]; }
  mu *= 0.01f;
  float var = 0.f;
  for (int o = 0; o < 100; ++o) { float d = s[o] - mu; var += d * d; }
  float istd = rsqrtf(var * 0.01f + 1e-5f);
  for (int o = 0; o < 100; ++o) {
    s[o] = (s[o] - mu) * istd * lng[o] + lnb[o];
    hs[(size_t)n * 100 + o] = s[o];
    aff0[(size_t)n * 160 + o] = f2bf(s[o]);
  }
  float v[16][3], msq = 0.f;
  for (int h = 0; h < 16; ++h) {
    float ss = 0.f;
    for (int c = 0; c < 3; ++c) {
      v[h][c] = hv[(size_t)n * 48 + h * 3 + c] + acc_v[(size_t)n * 48 + h * 3 + c] * inv;
      ss += v[h][c] * v[h][c];
    }
    msq += clipn(ss);
  }
  float vinv = rsqrtf(msq / 16.f);
  for (int h = 0; h < 16; ++h)
    for (int c = 0; c < 3; ++c) { v[h][c] *= vinv; hv[(size_t)n * 48 + h * 3 + c] = v[h][c]; }
  for (int h = 0; h < 32; ++h) {
    float x = 0, y = 0, z = 0;
    const float* w = whf0 + h * 16;
    for (int j = 0; j < 16; ++j) { float ww = w[j]; x += ww * v[j][0]; y += ww * v[j][1]; z += ww * v[j][2]; }
    vhf[(size_t)n * 96 + h * 3] = x; vhf[(size_t)n * 96 + h * 3 + 1] = y; vhf[(size_t)n * 96 + h * 3 + 2] = z;
    aff0[(size_t)n * 160 + 100 + h] = f2bf(sqrtf(clipn(x * x + y * y + z * z)));
  }
  for (int k = 132; k < 160; ++k) aff0[(size_t)n * 160 + k] = 0;
}

// ---------- generic row-GEMM with WMMA (A bf16 row-major in global) ----------
__global__ __launch_bounds__(32) void gemm_rows_kernel(const unsigned short* __restrict__ A,
                                                       int Kp, const unsigned short* __restrict__ Bp,
                                                       int KT, int NT, const float* __restrict__ bias,
                                                       int Nvalid, int mode,
                                                       unsigned short* __restrict__ dstB,
                                                       int dstStride, float* __restrict__ dstF,
                                                       int M) {
  int lane = threadIdx.x & 31;
  int mt = blockIdx.x;
  int m = mt * 16 + (lane & 15);
  const unsigned short* arow = A + (size_t)m * Kp;
  for (int nt = 0; nt < NT; ++nt) {
    v8f acc = zero8();
    for (int kt = 0; kt < KT; ++kt) {
      v16bf a = load_a_frag(arow, kt, lane);
      v16bf b = load_b_frag(Bp, kt, nt, NT, lane);
      acc = __builtin_amdgcn_wmma_f32_16x16x32_bf16(false, a, false, b, (short)0, acc, false, false);
    }
    int n = nt * 16 + (lane & 15);
    float bs = (n < Nvalid) ? bias[n] : 0.f;
#pragma unroll
    for (int r = 0; r < 8; ++r) {
      int mm = mt * 16 + r + ((lane & 16) ? 8 : 0);
      if (n < Nvalid && mm < M) {
        float val = acc[r] + bs;
        if (mode == 0) dstB[(size_t)mm * dstStride + n] = f2bf(fmaxf(val, 0.f));
        else           dstF[(size_t)mm * Nvalid + n] = val;
      }
    }
  }
}

// ---------- ff vector path ----------
__global__ void node_ffvec_kernel(const float* __restrict__ vhf, const float* __restrict__ wvf0,
                                  const float* __restrict__ whf1, const float* __restrict__ wvf1,
                                  unsigned short* __restrict__ aff1, float* __restrict__ acc_v,
                                  int N) {
  int n = blockIdx.x * blockDim.x + threadIdx.x;
  if (n >= N) return;
  float vh0[32][3];
  for (int h = 0; h < 32; ++h)
    for (int c = 0; c < 3; ++c) vh0[h][c] = vhf[(size_t)n * 96 + h * 3 + c];
  float vf[32][3];
  for (int o = 0; o < 32; ++o) {
    float x = 0, y = 0, z = 0;
    const float* w = wvf0 + o * 32;
    for (int h = 0; h < 32; ++h) { float ww = w[h]; x += ww * vh0[h][0]; y += ww * vh0[h][1]; z += ww * vh0[h][2]; }
    float g = sigm(sqrtf(clipn(x * x + y * y + z * z)));
    vf[o][0] = x * g; vf[o][1] = y * g; vf[o][2] = z * g;
  }
  float vh1[32][3];
  for (int h = 0; h < 32; ++h) {
    float x = 0, y = 0, z = 0;
    const float* w = whf1 + h * 32;
    for (int j = 0; j < 32; ++j) { float ww = w[j]; x += ww * vf[j][0]; y += ww * vf[j][1]; z += ww * vf[j][2]; }
    vh1[h][0] = x; vh1[h][1] = y; vh1[h][2] = z;
    aff1[(size_t)n * 448 + 400 + h] = f2bf(sqrtf(clipn(x * x + y * y + z * z)));
  }
  for (int k = 432; k < 448; ++k) aff1[(size_t)n * 448 + k] = 0;
  for (int o = 0; o < 16; ++o) {
    float x = 0, y = 0, z = 0;
    const float* w = wvf1 + o * 32;
    for (int h = 0; h < 32; ++h) { float ww = w[h]; x += ww * vh1[h][0]; y += ww * vh1[h][1]; z += ww * vh1[h][2]; }
    acc_v[(size_t)n * 48 + o * 3] = x;
    acc_v[(size_t)n * 48 + o * 3 + 1] = y;
    acc_v[(size_t)n * 48 + o * 3 + 2] = z;
  }
}

// ---------- residual + LN1 (and zero accumulators for next layer) ----------
__global__ void node_ln1_kernel(float* __restrict__ hs, float* __restrict__ hv,
                                float* __restrict__ acc_s, float* __restrict__ acc_v,
                                const float* __restrict__ lng, const float* __restrict__ lnb,
                                int N) {
  int n = blockIdx.x * blockDim.x + threadIdx.x;
  if (n >= N) return;
  float s[100], mu = 0.f;
  for (int o = 0; o < 100; ++o) {
    s[o] = hs[(size_t)n * 100 + o] + acc_s[(size_t)n * 100 + o];
    acc_s[(size_t)n * 100 + o] = 0.f;
    mu += s[o];
  }
  mu *= 0.01f;
  float var = 0.f;
  for (int o = 0; o < 100; ++o) { float d = s[o] - mu; var += d * d; }
  float istd = rsqrtf(var * 0.01f + 1e-5f);
  for (int o = 0; o < 100; ++o) hs[(size_t)n * 100 + o] = (s[o] - mu) * istd * lng[o] + lnb[o];
  float v[16][3], msq = 0.f;
  for (int h = 0; h < 16; ++h) {
    float ss = 0.f;
    for (int c = 0; c < 3; ++c) {
      v[h][c] = hv[(size_t)n * 48 + h * 3 + c] + acc_v[(size_t)n * 48 + h * 3 + c];
      acc_v[(size_t)n * 48 + h * 3 + c] = 0.f;
      ss += v[h][c] * v[h][c];
    }
    msq += clipn(ss);
  }
  float vinv = rsqrtf(msq / 16.f);
  for (int h = 0; h < 16; ++h)
    for (int c = 0; c < 3; ++c) hv[(size_t)n * 48 + h * 3 + c] = v[h][c] * vinv;
}

// ---------- output head ----------
__global__ void out_prep_kernel(const float* __restrict__ hs, const float* __restrict__ hv,
                                const float* __restrict__ lng, const float* __restrict__ lnb,
                                const float* __restrict__ wh, float* __restrict__ sinout, int N) {
  int n = blockIdx.x * blockDim.x + threadIdx.x;
  if (n >= N) return;
  float s[100], mu = 0.f;
  for (int o = 0; o < 100; ++o) { s[o] = hs[(size_t)n * 100 + o]; mu += s[o]; }
  mu *= 0.01f;
  float var = 0.f;
  for (int o = 0; o < 100; ++o) { float d = s[o] - mu; var += d * d; }
  float istd = rsqrtf(var * 0.01f + 1e-5f);
  for (int o = 0; o < 100; ++o) sinout[(size_t)n * 116 + o] = (s[o] - mu) * istd * lng[o] + lnb[o];
  float v[16][3], msq = 0.f;
  for (int h = 0; h < 16; ++h) {
    float ss = 0.f;
    for (int c = 0; c < 3; ++c) { v[h][c] = hv[(size_t)n * 48 + h * 3 + c]; ss += v[h][c] * v[h][c]; }
    msq += clipn(ss);
  }
  float vinv = rsqrtf(msq / 16.f);
  for (int h = 0; h < 16; ++h)
    for (int c = 0; c < 3; ++c) v[h][c] *= vinv;
  for (int h = 0; h < 16; ++h) {
    float x = 0, y = 0, z = 0;
    const float* w = wh + h * 16;
    for (int j = 0; j < 16; ++j) { float ww = w[j]; x += ww * v[j][0]; y += ww * v[j][1]; z += ww * v[j][2]; }
    sinout[(size_t)n * 116 + 100 + h] = sqrtf(clipn(x * x + y * y + z * z));
  }
}

__global__ void out_gemm_kernel(const float* __restrict__ sinout, const float* __restrict__ wsw,
                                const float* __restrict__ wsb, float* __restrict__ out, int N) {
  int idx = blockIdx.x * blockDim.x + threadIdx.x;
  if (idx >= N * 100) return;
  int n = idx / 100, o = idx - n * 100;
  const float* s = sinout + (size_t)n * 116;
  const float* w = wsw + o * 116;
  float a = wsb[o];
  for (int k = 0; k < 116; ++k) a += w[k] * s[k];
  out[idx] = fmaxf(a, 0.f);
}

// =================================================================================
extern "C" void kernel_launch(void* const* d_in, const int* in_sizes, int n_in, void* d_out,
                              int out_size, void* d_ws, size_t ws_size, hipStream_t stream) {
  (void)in_sizes; (void)n_in; (void)out_size; (void)ws_size;
  const float* node_s = (const float*)d_in[0];
  const float* node_v = (const float*)d_in[1];
  const float* edge_s = (const float*)d_in[2];
  const float* edge_v = (const float*)d_in[3];
  const int* eidx = (const int*)d_in[4];
  auto P = [&](int i) { return (const float*)d_in[i]; };
  // param leaf index map (jax pytree: dict keys sorted):
  // 5..8  We_gvp {wh, ws_b, ws_w, wv}; 9,10 We_ln {b,g}
  // 11..13 Wout_gvp {wh, ws_b, ws_w}; 14,15 Wout_ln {b,g}
  // 16..19 Wv_gvp {wh, ws_b, ws_w, wv}; 20,21 Wv_ln {b,g}
  // per layer l (base lb = 22 + 24*l):
  //   ff0: wh lb+0, ws_b lb+1, ws_w lb+2, wv lb+3 | ff1: lb+4..7
  //   ln0: b lb+8, g lb+9 | ln1: b lb+10, g lb+11
  //   msg0: wh lb+12, ws_b lb+13, ws_w lb+14, wv lb+15 | msg1: lb+16..19 | msg2: lb+20..23

  size_t off = 0;
  auto take = [&](size_t bytes) -> char* {
    char* p = (char*)d_ws + off;
    off += (bytes + 255) & ~(size_t)255;
    return p;
  };
  float* hs = (float*)take((size_t)Nn * 100 * 4);
  float* hv = (float*)take((size_t)Nn * 48 * 4);
  float* es2 = (float*)take((size_t)Ne * 32 * 4);
  float* ev2 = (float*)take((size_t)Ne * 3 * 4);
  float* acc_s = (float*)take((size_t)Nn * 100 * 4);
  float* acc_v = (float*)take((size_t)Nn * 48 * 4);
  float* invdeg = (float*)take((size_t)Nn * 4);
  unsigned int* degcnt = (unsigned int*)take((size_t)Nn * 4);
  float* vhf = (float*)take((size_t)Nn * 96 * 4);
  unsigned short* aff0 = (unsigned short*)take((size_t)Nn * 160 * 2);
  unsigned short* aff1 = (unsigned short*)take((size_t)Nn * 448 * 2);
  float* sinout = (float*)take((size_t)Nn * 116 * 4);
  unsigned short *pmsg0[3], *pmsg1[3], *pmsg2[3], *pff0[3], *pff1[3];
  for (int l = 0; l < 3; ++l) {
    pmsg0[l] = (unsigned short*)take((size_t)9 * 7 * 512 * 2);
    pmsg1[l] = (unsigned short*)take((size_t)4 * 7 * 512 * 2);
    pmsg2[l] = (unsigned short*)take((size_t)4 * 7 * 512 * 2);
    pff0[l] = (unsigned short*)take((size_t)5 * 25 * 512 * 2);
    pff1[l] = (unsigned short*)take((size_t)14 * 7 * 512 * 2);
  }

  hipMemsetAsync(acc_s, 0, (size_t)Nn * 100 * 4, stream);
  hipMemsetAsync(acc_v, 0, (size_t)Nn * 48 * 4, stream);
  hipMemsetAsync(degcnt, 0, (size_t)Nn * 4, stream);

  for (int l = 0; l < 3; ++l) {
    int lb = 22 + l * 24;
    pack_b_kernel<<<64, 256, 0, stream>>>(P(lb + 14), 100, 265, 9, 7, pmsg0[l]);
    pack_b_kernel<<<64, 256, 0, stream>>>(P(lb + 18), 100, 116, 4, 7, pmsg1[l]);
    pack_b_kernel<<<64, 256, 0, stream>>>(P(lb + 22), 100, 116, 4, 7, pmsg2[l]);
    pack_b_kernel<<<64, 256, 0, stream>>>(P(lb + 2), 400, 132, 5, 25, pff0[l]);
    pack_b_kernel<<<64, 256, 0, stream>>>(P(lb + 6), 100, 432, 14, 7, pff1[l]);
  }
  deg_count_kernel<<<(Ne + 255) / 256, 256, 0, stream>>>(eidx, degcnt, Ne);
  deg_inv_kernel<<<(Nn + 255) / 256, 256, 0, stream>>>(degcnt, invdeg, Nn);
  embed_nodes_kernel<<<(Nn + 127) / 128, 128, 0, stream>>>(node_s, node_v, P(21), P(20), P(16),
                                                           P(17), P(18), P(19), hs, hv, Nn);
  embed_edges_kernel<<<(Ne + 255) / 256, 256, 0, stream>>>(edge_s, edge_v, P(9), P(10), P(5),
                                                           P(6), P(7), P(8), es2, ev2, Ne);

  for (int l = 0; l < 3; ++l) {
    int lb = 22 + l * 24;
    edge_msg_kernel<<<Ne / 32, 64, 0, stream>>>(
        hs, hv, es2, ev2, eidx, P(lb + 12), P(lb + 15), P(lb + 16), P(lb + 19), P(lb + 20),
        P(lb + 23), P(lb + 13), P(lb + 17), P(lb + 21), pmsg0[l], pmsg1[l], pmsg2[l], acc_s,
        acc_v, Ne);
    node_ln0_kernel<<<(Nn + 127) / 128, 128, 0, stream>>>(acc_s, acc_v, invdeg, P(lb + 9),
                                                          P(lb + 8), P(lb + 0), hs, hv, vhf,
                                                          aff0, Nn);
    gemm_rows_kernel<<<Nn / 16, 32, 0, stream>>>(aff0, 160, pff0[l], 5, 25, P(lb + 1), 400, 0,
                                                 aff1, 448, nullptr, Nn);
    node_ffvec_kernel<<<(Nn + 127) / 128, 128, 0, stream>>>(vhf, P(lb + 3), P(lb + 4), P(lb + 7),
                                                            aff1, acc_v, Nn);
    gemm_rows_kernel<<<Nn / 16, 32, 0, stream>>>(aff1, 448, pff1[l], 14, 7, P(lb + 5), 100, 1,
                                                 nullptr, 0, acc_s, Nn);
    node_ln1_kernel<<<(Nn + 127) / 128, 128, 0, stream>>>(hs, hv, acc_s, acc_v, P(lb + 11),
                                                          P(lb + 10), Nn);
  }
  out_prep_kernel<<<(Nn + 127) / 128, 128, 0, stream>>>(hs, hv, P(15), P(14), P(11), sinout, Nn);
  out_gemm_kernel<<<(Nn * 100 + 255) / 256, 256, 0, stream>>>(sinout, P(13), P(12),
                                                              (float*)d_out, Nn);
}
